// FullAttentionLayer_82386062672326
// MI455X (gfx1250) — matrix operand order repaired and checked by reading, not compiled, not run on hardware
//
#include <hip/hip_runtime.h>
#include <stdint.h>

// ---------------- types / helpers ----------------
typedef __attribute__((ext_vector_type(16))) __bf16 v16bf;
typedef __attribute__((ext_vector_type(8)))  float  v8f;

union FragB { uint4 u[2]; v16bf v; };   // 32 bytes = 16 bf16 = one WMMA A/B operand

__device__ __forceinline__ unsigned short f2bf(float f) {
    unsigned int u = __float_as_uint(f);
    u += 0x7FFFu + ((u >> 16) & 1u);       // round-to-nearest-even
    return (unsigned short)(u >> 16);
}

__device__ __forceinline__ v8f wmma_bf16(const FragB& a, const FragB& b, v8f c) {
    return __builtin_amdgcn_wmma_f32_16x16x32_bf16(
        /*neg_a=*/false, a.v, /*neg_b=*/false, b.v,
        /*c_mod=*/(short)0, c, /*reuse_a=*/false, /*reuse_b=*/false);
}

// LDS byte offset of a __shared__ object: low 32 bits of the generic address
__device__ __forceinline__ unsigned lds_off(const void* p) {
    return (unsigned)(unsigned long long)p;
}

// CDNA5 async copy: global -> LDS, 16B per lane, tracked by ASYNCcnt.
// SADDR form: vdst = LDS byte address, voff = 32-bit byte offset, saddr = base.
__device__ __forceinline__ void async_g2l_b128(unsigned ldsAddr, unsigned voff,
                                               const void* base) {
    asm volatile("global_load_async_to_lds_b128 %0, %1, %2"
                 :: "v"(ldsAddr), "v"(voff), "s"(base) : "memory");
}
__device__ __forceinline__ void wait_async0() {
    asm volatile("s_wait_asynccnt 0x0" ::: "memory");
}

#define S_LEN 4096
#define D_DIM 1024
#define NHEAD 16
#define HDIM  64

// ---------------- fp32 -> bf16 convert ----------------
__global__ void k_cvt_bf16(const float* __restrict__ in,
                           unsigned short* __restrict__ out, int n) {
    int i = (blockIdx.x * blockDim.x + threadIdx.x) * 4;
    if (i >= n) return;
    float4 v = *(const float4*)(in + i);
    uint2 o;
    o.x = (unsigned)f2bf(v.x) | ((unsigned)f2bf(v.y) << 16);
    o.y = (unsigned)f2bf(v.z) | ((unsigned)f2bf(v.w) << 16);
    *(uint2*)(out + i) = o;
}

// out_w is [e][d]; store Wt[d][e] = out_w[e][d] as bf16 (torch Linear: x @ W^T)
__global__ void k_transpose_bf16(const float* __restrict__ in,
                                 unsigned short* __restrict__ out) {
    __shared__ float t[32][33];
    int bx = blockIdx.x * 32, by = blockIdx.y * 32;
    int tx = threadIdx.x, ty = threadIdx.y;        // 32 x 8
    for (int i = 0; i < 32; i += 8)
        t[ty + i][tx] = in[(size_t)(by + ty + i) * D_DIM + bx + tx];
    __syncthreads();
    for (int i = 0; i < 32; i += 8)
        out[(size_t)(bx + ty + i) * D_DIM + by + tx] = f2bf(t[tx][ty + i]);
}

// ---------------- bf16 WMMA GEMM: C[4096x1024] = A[4096x1024] * B[1024x1024] ----
// mode: -1 -> use blockIdx.z as mode {0:Q, 1:K^T, 2:V}; 3 -> plain fp32 out
#define BM 128
#define BN 128
#define BK 32
#define LDA_S 48    // LDS row stride (u16) for A tile, 96B = 16B multiple
#define LDB_S 136   // LDS row stride (u16) for B tile, 272B = 16B multiple

__global__ __launch_bounds__(256) void k_gemm_bf16(
    const unsigned short* __restrict__ A,
    const unsigned short* __restrict__ Wbase, int wstride, int mode,
    float* __restrict__ outF,
    unsigned short* __restrict__ outQ,
    unsigned short* __restrict__ outK,
    unsigned short* __restrict__ outV)
{
    const int N = D_DIM, K = D_DIM;
    int z = blockIdx.z;
    const unsigned short* B = Wbase + (size_t)z * wstride;
    int md = (mode < 0) ? z : mode;

    __shared__ __align__(16) unsigned short As[2][BM * LDA_S];
    __shared__ __align__(16) unsigned short Bs[2][BK * LDB_S];

    int tid = threadIdx.x;
    int lane = tid & 31, wave = tid >> 5;
    int wm = wave >> 1, wn = wave & 1;          // 4x2 wave grid -> 32x64 per wave
    int mBase = blockIdx.y * BM;
    int nBase = blockIdx.x * BN;
    int half = lane >> 4, mrow = lane & 15;

    v8f acc[2][4];
    for (int i = 0; i < 2; i++)
        for (int j = 0; j < 4; j++)
            for (int e = 0; e < 8; e++) acc[i][j][e] = 0.0f;

    // async staging: 512 x 16B chunks per tile, 2 per thread per matrix
    auto prefetch = [&](int kb, int buf) {
        #pragma unroll
        for (int i = 0; i < 2; i++) {
            int c = tid * 2 + i;                 // 0..511
            int r = c >> 2, p = c & 3;           // A: 128 rows x 4 chunks
            async_g2l_b128(lds_off(&As[buf][r * LDA_S + p * 8]),
                           (unsigned)(((mBase + r) * K + kb * BK + p * 8) * 2), A);
            int rr = c >> 4, cc = c & 15;        // B: 32 rows x 16 chunks
            async_g2l_b128(lds_off(&Bs[buf][rr * LDB_S + cc * 8]),
                           (unsigned)(((kb * BK + rr) * N + nBase + cc * 8) * 2), B);
        }
    };

    prefetch(0, 0);
    wait_async0();
    __syncthreads();

    const int nk = K / BK;                       // 32
    for (int kb = 0; kb < nk; ++kb) {
        int buf = kb & 1;
        if (kb + 1 < nk) prefetch(kb + 1, buf ^ 1);   // async engine fills next buf

        FragB af[2], bfg[4];
        #pragma unroll
        for (int i = 0; i < 2; i++) {            // A frags: 2 m-tiles
            const unsigned short* p = &As[buf][(wm * 32 + i * 16 + mrow) * LDA_S];
            af[i].u[0] = *(const uint4*)(p + half * 8);
            af[i].u[1] = *(const uint4*)(p + 16 + half * 8);
        }
        #pragma unroll
        for (int j = 0; j < 4; j++) {            // B frags: 4 n-tiles
            const unsigned short* p = &Bs[buf][wn * 64 + j * 16 + mrow];
            for (int s = 0; s < 16; s++)
                ((unsigned short*)&bfg[j])[s] = p[(half * 8 + (s & 7) + (s >> 3) * 16) * LDB_S];
        }
        #pragma unroll
        for (int i = 0; i < 2; i++)
            #pragma unroll
            for (int j = 0; j < 4; j++)
                acc[i][j] = wmma_bf16(af[i], bfg[j], acc[i][j]);

        wait_async0();                            // own prefetch done
        __syncthreads();                          // everyone's prefetch done
    }

    // epilogue
    #pragma unroll
    for (int i = 0; i < 2; i++) {
        #pragma unroll
        for (int j = 0; j < 4; j++) {
            int nCol = nBase + wn * 64 + j * 16 + mrow;
            #pragma unroll
            for (int r = 0; r < 8; r++) {
                int mRow = mBase + wm * 32 + i * 16 + r + 8 * half;
                float val = acc[i][j][r];
                if (md == 3) {
                    outF[(size_t)mRow * N + nCol] = val;
                } else {
                    int h = nCol >> 6, d = nCol & 63;
                    unsigned short bv = f2bf(val);
                    if (md == 0)       outQ[((size_t)h * S_LEN + mRow) * HDIM + d] = bv;
                    else if (md == 1)  outK[((size_t)h * HDIM + d) * S_LEN + mRow] = bv;
                    else               outV[((size_t)h * S_LEN + mRow) * HDIM + d] = bv;
                }
            }
        }
    }
}

// ---------------- flash attention (bf16 WMMA, fp32 softmax) ----------------
// grid: (S/64, H), block: 128 (4 waves x 16 query rows)
#define LDK_S 72   // 144B rows, 16B multiple

__global__ __launch_bounds__(128) void k_attn(
    const unsigned short* __restrict__ Q,   // [H][S][64]
    const unsigned short* __restrict__ Kt,  // [H][64][S]
    const unsigned short* __restrict__ V,   // [H][S][64]
    unsigned short* __restrict__ ctxOut)    // [S][1024]
{
    int h = blockIdx.y;
    int qb = blockIdx.x;
    int tid = threadIdx.x, lane = tid & 31, wave = tid >> 5;
    int half = lane >> 4, mrow = lane & 15;

    __shared__ __align__(16) unsigned short Ks[64 * LDK_S];
    __shared__ __align__(16) unsigned short Vs[64 * LDK_S];
    __shared__ __align__(16) unsigned short Ps[4][16 * LDK_S];

    // Q fragments for this wave's 16 rows (K=0..31 and K=32..63)
    FragB aq[2];
    {
        int qrow = qb * 64 + wave * 16 + mrow;
        const unsigned short* qp = Q + ((size_t)h * S_LEN + qrow) * HDIM;
        aq[0].u[0] = *(const uint4*)(qp + half * 8);
        aq[0].u[1] = *(const uint4*)(qp + 16 + half * 8);
        aq[1].u[0] = *(const uint4*)(qp + 32 + half * 8);
        aq[1].u[1] = *(const uint4*)(qp + 48 + half * 8);
    }

    v8f ctx[4];
    for (int j = 0; j < 4; j++)
        for (int e = 0; e < 8; e++) ctx[j][e] = 0.0f;
    float mrun[8], lrun[8];
    for (int r = 0; r < 8; r++) { mrun[r] = -1e30f; lrun[r] = 0.0f; }

    const unsigned short* KtB = Kt + (size_t)h * HDIM * S_LEN;
    const unsigned short* VB  = V  + (size_t)h * S_LEN * HDIM;

    for (int n0 = 0; n0 < S_LEN; n0 += 64) {
        __syncthreads();                        // previous block's LDS reads done
        #pragma unroll
        for (int i = 0; i < 4; i++) {           // async stage K^T and V blocks
            int c = tid * 4 + i;                // 0..511
            int row = c >> 3, p = c & 7;
            async_g2l_b128(lds_off(&Ks[row * LDK_S + p * 8]),
                           (unsigned)((row * S_LEN + n0 + p * 8) * 2), KtB);
            async_g2l_b128(lds_off(&Vs[row * LDK_S + p * 8]),
                           (unsigned)(((n0 + row) * HDIM + p * 8) * 2), VB);
        }
        wait_async0();
        __syncthreads();

        // scores: 16 x 64 tile per wave, contraction over Hd=64 (2 k-steps)
        v8f sc[4];
        for (int j = 0; j < 4; j++)
            for (int e = 0; e < 8; e++) sc[j][e] = 0.0f;
        #pragma unroll
        for (int kk = 0; kk < 2; kk++) {
            const unsigned short* kp = &Ks[(kk * 32) * LDK_S + mrow];
            #pragma unroll
            for (int j = 0; j < 4; j++) {
                FragB bk;
                for (int s = 0; s < 16; s++)
                    ((unsigned short*)&bk)[s] = kp[(half * 8 + (s & 7) + (s >> 3) * 16) * LDK_S + j * 16];
                sc[j] = wmma_bf16(aq[kk], bk, sc[j]);
            }
        }
        #pragma unroll
        for (int j = 0; j < 4; j++)
            for (int r = 0; r < 8; r++) sc[j][r] *= 0.125f;   // 1/sqrt(64)

        // online softmax (row = r + 8*half, spread over a 16-lane group)
        float newm[8], corr[8], psum[8];
        #pragma unroll
        for (int r = 0; r < 8; r++) {
            float mx = fmaxf(fmaxf(sc[0][r], sc[1][r]), fmaxf(sc[2][r], sc[3][r]));
            for (int off = 1; off < 16; off <<= 1)
                mx = fmaxf(mx, __shfl_xor(mx, off, 32));
            newm[r] = fmaxf(mrun[r], mx);
            corr[r] = __expf(mrun[r] - newm[r]);
            psum[r] = 0.0f;
        }
        #pragma unroll
        for (int j = 0; j < 4; j++)
            for (int r = 0; r < 8; r++) {
                float p = __expf(sc[j][r] - newm[r]);
                sc[j][r] = p;
                psum[r] += p;
            }
        #pragma unroll
        for (int r = 0; r < 8; r++) {
            for (int off = 1; off < 16; off <<= 1)
                psum[r] += __shfl_xor(psum[r], off, 32);
            lrun[r] = lrun[r] * corr[r] + psum[r];
            mrun[r] = newm[r];
        }
        #pragma unroll
        for (int j = 0; j < 4; j++)
            for (int r = 0; r < 8; r++) ctx[j][r] *= corr[r];

        // re-layout P: C-frag -> LDS -> A-frag (wave-private region, no barrier)
        unsigned short* pw = Ps[wave];
        #pragma unroll
        for (int j = 0; j < 4; j++)
            for (int r = 0; r < 8; r++)
                pw[(r + 8 * half) * LDK_S + j * 16 + mrow] = f2bf(sc[j][r]);

        FragB ap[2];
        #pragma unroll
        for (int kk = 0; kk < 2; kk++) {
            const unsigned short* pp = pw + mrow * LDK_S + kk * 32;
            ap[kk].u[0] = *(const uint4*)(pp + half * 8);
            ap[kk].u[1] = *(const uint4*)(pp + 16 + half * 8);
        }
        #pragma unroll
        for (int kk = 0; kk < 2; kk++) {
            const unsigned short* vp = &Vs[(kk * 32) * LDK_S + mrow];
            #pragma unroll
            for (int j = 0; j < 4; j++) {
                FragB bv;
                for (int s = 0; s < 16; s++)
                    ((unsigned short*)&bv)[s] = vp[(half * 8 + (s & 7) + (s >> 3) * 16) * LDK_S + j * 16];
                ctx[j] = wmma_bf16(ap[kk], bv, ctx[j]);
            }
        }
    }

    // finalize: ctx / l, scatter to [s][h*64 + d] bf16
    int qrow = qb * 64 + wave * 16;
    #pragma unroll
    for (int j = 0; j < 4; j++)
        for (int r = 0; r < 8; r++) {
            int row = qrow + r + 8 * half;
            int col = h * HDIM + j * 16 + mrow;
            ctxOut[(size_t)row * D_DIM + col] = f2bf(ctx[j][r] / lrun[r]);
        }
}

// ---------------- residual + LayerNorm (+bias, +ReLU) ----------------
__global__ __launch_bounds__(256) void k_ln(
    const float* __restrict__ a, const float* __restrict__ b,
    const float* __restrict__ bias,
    const float* __restrict__ g, const float* __restrict__ beta,
    float* __restrict__ outF, unsigned short* __restrict__ outB, int relu)
{
    int row = blockIdx.x, tid = threadIdx.x;
    __shared__ float red[256];
    float v[4];
    #pragma unroll
    for (int i = 0; i < 4; i++) {
        int c = tid + i * 256;
        float x = a[(size_t)row * D_DIM + c] + b[(size_t)row * D_DIM + c];
        if (bias) x += bias[c];
        v[i] = x;
    }
    float s = v[0] + v[1] + v[2] + v[3];
    red[tid] = s; __syncthreads();
    for (int st = 128; st > 0; st >>= 1) {
        if (tid < st) red[tid] += red[tid + st];
        __syncthreads();
    }
    float mu = red[0] * (1.0f / D_DIM);
    __syncthreads();
    float q = 0.0f;
    #pragma unroll
    for (int i = 0; i < 4; i++) { float d = v[i] - mu; q += d * d; }
    red[tid] = q; __syncthreads();
    for (int st = 128; st > 0; st >>= 1) {
        if (tid < st) red[tid] += red[tid + st];
        __syncthreads();
    }
    float inv = rsqrtf(red[0] * (1.0f / D_DIM) + 1e-5f);
    #pragma unroll
    for (int i = 0; i < 4; i++) {
        int c = tid + i * 256;
        float y = (v[i] - mu) * inv * g[c] + beta[c];
        if (relu) y = fmaxf(y, 0.0f);
        outF[(size_t)row * D_DIM + c] = y;
        if (outB) outB[(size_t)row * D_DIM + c] = f2bf(y);
    }
}

// ---------------- host orchestration ----------------
extern "C" void kernel_launch(void* const* d_in, const int* in_sizes, int n_in,
                              void* d_out, int out_size, void* d_ws, size_t ws_size,
                              hipStream_t stream) {
    const float* x        = (const float*)d_in[0];
    const float* W_q      = (const float*)d_in[1];
    const float* W_k      = (const float*)d_in[2];
    const float* W_v      = (const float*)d_in[3];
    const float* W_o      = (const float*)d_in[4];
    const float* out_w    = (const float*)d_in[5];
    const float* out_b    = (const float*)d_in[6];
    const float* ln_a_g   = (const float*)d_in[7];
    const float* ln_a_b   = (const float*)d_in[8];
    const float* ln_l_g   = (const float*)d_in[9];
    const float* ln_l_b   = (const float*)d_in[10];

    const size_t WMAT = (size_t)D_DIM * D_DIM;     // 1M
    const size_t SD   = (size_t)S_LEN * D_DIM;     // 4M

    char* ws = (char*)d_ws;
    size_t off = 0;
    auto alloc = [&](size_t bytes) -> void* {
        void* p = ws + off;
        off += (bytes + 255) & ~(size_t)255;
        return p;
    };
    // Wq/Wk/Wv must be contiguous (z-indexed in QKV GEMM): 2MB each, 256B-aligned
    unsigned short* Wqb  = (unsigned short*)alloc(WMAT * 2);
    unsigned short* Wkb  = (unsigned short*)alloc(WMAT * 2);
    unsigned short* Wvb  = (unsigned short*)alloc(WMAT * 2);
    unsigned short* Wob  = (unsigned short*)alloc(WMAT * 2);
    unsigned short* Wtb  = (unsigned short*)alloc(WMAT * 2);  // out_w^T
    unsigned short* Xb   = (unsigned short*)alloc(SD * 2);
    unsigned short* Qb   = (unsigned short*)alloc(SD * 2);    // [H][S][64]
    unsigned short* Ktb  = (unsigned short*)alloc(SD * 2);    // [H][64][S]
    unsigned short* Vb   = (unsigned short*)alloc(SD * 2);    // [H][S][64]
    unsigned short* Ctxb = (unsigned short*)alloc(SD * 2);    // [S][D]
    float*          XO   = (float*)alloc(SD * 4);             // proj out, reused for lin
    float*          MHA  = (float*)alloc(SD * 4);
    unsigned short* MHAb = (unsigned short*)alloc(SD * 2);

    // 1) convert activations + weights to bf16
    k_cvt_bf16<<<(int)(SD / 4 / 256), 256, 0, stream>>>(x, Xb, (int)SD);
    k_cvt_bf16<<<(int)(WMAT / 4 / 256), 256, 0, stream>>>(W_q, Wqb, (int)WMAT);
    k_cvt_bf16<<<(int)(WMAT / 4 / 256), 256, 0, stream>>>(W_k, Wkb, (int)WMAT);
    k_cvt_bf16<<<(int)(WMAT / 4 / 256), 256, 0, stream>>>(W_v, Wvb, (int)WMAT);
    k_cvt_bf16<<<(int)(WMAT / 4 / 256), 256, 0, stream>>>(W_o, Wob, (int)WMAT);
    k_transpose_bf16<<<dim3(32, 32), dim3(32, 8), 0, stream>>>(out_w, Wtb);

    // 2) fused QKV GEMMs (z = 0/1/2 selects weight + output routing)
    k_gemm_bf16<<<dim3(8, 32, 3), 256, 0, stream>>>(
        Xb, Wqb, (int)WMAT, -1, nullptr, Qb, Ktb, Vb);

    // 3) flash attention
    k_attn<<<dim3(S_LEN / 64, NHEAD), 128, 0, stream>>>(Qb, Ktb, Vb, Ctxb);

    // 4) attention output projection -> fp32
    k_gemm_bf16<<<dim3(8, 32, 1), 256, 0, stream>>>(
        Ctxb, Wob, 0, 3, XO, nullptr, nullptr, nullptr);

    // 5) LN1: mha = LN(x + proj) -> fp32 + bf16 copies
    k_ln<<<S_LEN, 256, 0, stream>>>(XO, x, nullptr, ln_a_g, ln_a_b, MHA, MHAb, 0);

    // 6) output linear: lin = mha @ out_w^T  (bias folded into LN2), reuse XO
    k_gemm_bf16<<<dim3(8, 32, 1), 256, 0, stream>>>(
        MHAb, Wtb, 0, 3, XO, nullptr, nullptr, nullptr);

    // 7) LN2 + ReLU -> d_out
    k_ln<<<S_LEN, 256, 0, stream>>>(XO, MHA, out_b, ln_l_g, ln_l_b,
                                    (float*)d_out, nullptr, 1);
}
